// MultiHeadCrossAttention_3083786519235
// MI455X (gfx1250) — matrix operand order repaired and checked by reading, not compiled, not run on hardware
//
#include <hip/hip_runtime.h>
#include <hip/hip_bf16.h>

typedef __attribute__((ext_vector_type(16))) __bf16 v16bf;
typedef __attribute__((ext_vector_type(8)))  __bf16 v8bf;
typedef __attribute__((ext_vector_type(8)))  float  v8f;

#define BQ 8
#define HQ 16
#define LQ 1024
#define DMODEL 1024
#define DEPTH 64

union V16U { v16bf v; v8bf half8[2]; };

static __device__ __forceinline__ v8f wmma_bf16(v16bf a, v16bf b, v8f c) {
    return __builtin_amdgcn_wmma_f32_16x16x32_bf16(
        /*neg_a=*/false, a, /*neg_b=*/false, b,
        /*c_mod=*/(short)0, c, /*reuse_a=*/false, /*reuse_b=*/false);
}

// ---------------------------------------------------------------------------
// f32 -> bf16 elementwise convert (weights)
// ---------------------------------------------------------------------------
__global__ void cvt_bf16_kernel(const float* __restrict__ in,
                                __bf16* __restrict__ out, int n) {
    int i = blockIdx.x * blockDim.x + threadIdx.x;
    if (i < n) out[i] = (__bf16)in[i];
}

// ---------------------------------------------------------------------------
// Per-head projection: out[b,h] = X[b] (L x 1024) @ W[h]^T (1024 x 64), bf16.
// Block = 128 threads (4 waves). Tile: 64 rows x 64 depth, K-loop of 32.
// transposed==0 : out is [B*H, L, 64]   (row-major, used for Q and K)
// transposed==1 : out is [B*H, 64, L]   (V^T, so attention PV B-loads are rows)
// Epilogue stages the 64x64 bf16 tile in LDS, then writes coalesced b128.
// ---------------------------------------------------------------------------
__global__ void proj_kernel(const float* __restrict__ X,
                            const __bf16* __restrict__ W,   // [H, 64, 1024] bf16
                            __bf16* __restrict__ out,
                            int transposed) {
    __shared__ __attribute__((aligned(16))) __bf16 stile[64][72];

    const int tid  = threadIdx.x;
    const int w    = tid >> 5;           // wave 0..3 -> rows [16w, 16w+16)
    const int lane = tid & 31;
    const int ln   = lane & 15;
    const int hi   = lane >> 4;          // half-wave select
    const int bh   = blockIdx.y;         // b*H + h
    const int hd   = bh & (HQ - 1);
    const int b    = bh >> 4;
    const int mbase = blockIdx.x * 64;

    const float*  Xrow = X + (size_t)b * LQ * DMODEL
                           + (size_t)(mbase + w * 16 + ln) * DMODEL;
    const __bf16* Wh   = W + (size_t)hd * DEPTH * DMODEL;

    v8f acc[4] = {};

    for (int k = 0; k < DMODEL; k += 32) {
        // A tile, 16x32 bf16 A-layout: elems 0..7 -> K=hi*8+j, 8..15 -> 16+hi*8+j
        v16bf a;
        const float* p0 = Xrow + k + hi * 8;
        const float* p1 = Xrow + k + 16 + hi * 8;
#pragma unroll
        for (int j = 0; j < 8; ++j) {
            a[j]     = (__bf16)p0[j];
            a[8 + j] = (__bf16)p1[j];
        }
#pragma unroll
        for (int nt = 0; nt < 4; ++nt) {
            // B 32x16: lane = column n (= output depth e), elems 0..7 -> K=hi*8+j, 8..15 -> 16+hi*8+j
            const __bf16* wp = Wh + (size_t)(nt * 16 + ln) * DMODEL + k + hi * 8;
            V16U wb;
            wb.half8[0] = *(const v8bf*)wp;
            wb.half8[1] = *(const v8bf*)(wp + 16);
            v16bf bm = wb.v;
            acc[nt] = wmma_bf16(a, bm, acc[nt]);
        }
    }

    // ---- stage C tile in LDS in the *output* layout (uniform branch) ----
    // C layout: lanes 0-15 hold M=r (VGPR r), lanes 16-31 hold M=8+r; N = ln.
    if (!transposed) {
#pragma unroll
        for (int nt = 0; nt < 4; ++nt)
#pragma unroll
            for (int r = 0; r < 8; ++r)
                stile[w * 16 + hi * 8 + r][nt * 16 + ln] = (__bf16)acc[nt][r];
    } else {
#pragma unroll
        for (int nt = 0; nt < 4; ++nt) {
            v8bf pk;
#pragma unroll
            for (int r = 0; r < 8; ++r) pk[r] = (__bf16)acc[nt][r];
            *(v8bf*)&stile[nt * 16 + ln][w * 16 + hi * 8] = pk;   // contiguous rows
        }
    }
    __syncthreads();

    // ---- coalesced write-back: 512 chunks of 16B, 4 per thread ----
    const size_t obase_n = ((size_t)bh * LQ + mbase) * DEPTH;  // + row*64
    const size_t obase_t = (size_t)bh * DEPTH * LQ + mbase;    // + row*1024
#pragma unroll
    for (int i = 0; i < 4; ++i) {
        int c   = tid + i * 128;
        int row = c >> 3;
        int c8  = c & 7;
        v8bf vv = *(const v8bf*)&stile[row][c8 * 8];
        size_t rowbase = transposed ? (obase_t + (size_t)row * LQ)
                                    : (obase_n + (size_t)row * DEPTH);
        *(v8bf*)(out + rowbase + c8 * 8) = vv;
    }
}

// ---------------------------------------------------------------------------
// Flash attention: per block (64 Q rows, one (b,h)); 4 waves x 16 rows.
// K/V tiles are double-buffered in LDS via global_load_async_to_lds_b128
// (ASYNCcnt), so all 4 waves share one copy of each tile.
// Qb,Kb: [B*H, L, 64] bf16 ; Vt: [B*H, 64, L] bf16 ; out: [B, L, H, 64] f32
// ---------------------------------------------------------------------------
__global__ void attn_kernel(const __bf16* __restrict__ Qb,
                            const __bf16* __restrict__ Kb,
                            const __bf16* __restrict__ Vt,
                            float* __restrict__ out) {
    __shared__ __attribute__((aligned(16))) __bf16 Klds[2][64][72];
    __shared__ __attribute__((aligned(16))) __bf16 Vlds[2][64][72];
    __shared__ __attribute__((aligned(16))) __bf16 pbuf[4][16][72]; // wave-private P
    __shared__ __attribute__((aligned(16))) float  otile[64][68];

    const int tid  = threadIdx.x;
    const int w    = tid >> 5;
    const int lane = tid & 31;
    const int ln   = lane & 15;
    const int hi   = lane >> 4;
    const int bh   = blockIdx.y;
    const int hd   = bh & (HQ - 1);
    const int b    = bh >> 4;
    const int mbase = blockIdx.x * 64;

    const __bf16* Kbase = Kb + (size_t)bh * LQ * DEPTH;
    const __bf16* Vbase = Vt + (size_t)bh * DEPTH * LQ;

    // ---- async stage of one K tile (64x64) + V tile (64x64) into LDS ----
    auto issue_tile = [&](int kt, int buf) {
#pragma unroll
        for (int i = 0; i < 4; ++i) {
            int c   = tid + i * 128;    // 512 chunks of 16B per tensor
            int row = c >> 3;
            int c8  = c & 7;
            const __bf16* gk = Kbase + (size_t)(kt + row) * DEPTH + c8 * 8;
            unsigned lk = (unsigned)(uintptr_t)(const void*)&Klds[buf][row][c8 * 8];
            asm volatile("global_load_async_to_lds_b128 %0, %1, off"
                         :: "v"(lk), "v"(gk) : "memory");
            const __bf16* gv = Vbase + (size_t)row * LQ + kt + c8 * 8;
            unsigned lv = (unsigned)(uintptr_t)(const void*)&Vlds[buf][row][c8 * 8];
            asm volatile("global_load_async_to_lds_b128 %0, %1, off"
                         :: "v"(lv), "v"(gv) : "memory");
        }
    };

    // Q rows for this wave, held resident in A-layout for the whole K loop.
    const __bf16* Qp = Qb + ((size_t)bh * LQ + mbase + w * 16 + ln) * DEPTH;
    V16U t;
    v16bf qa0, qa1;
    t.half8[0] = *(const v8bf*)(Qp + hi * 8);
    t.half8[1] = *(const v8bf*)(Qp + 16 + hi * 8);
    qa0 = t.v;
    t.half8[0] = *(const v8bf*)(Qp + 32 + hi * 8);
    t.half8[1] = *(const v8bf*)(Qp + 48 + hi * 8);
    qa1 = t.v;

    v8f o[4] = {};
    float mrow[8], lrow[8];
#pragma unroll
    for (int r = 0; r < 8; ++r) { mrow[r] = -1e30f; lrow[r] = 0.0f; }
    const float scale = 0.125f;   // 1/sqrt(64)

    issue_tile(0, 0);

    for (int it = 0; it < LQ / 64; ++it) {
        const int buf = it & 1;
        asm volatile("s_wait_asynccnt 0x0" ::: "memory");
        __syncthreads();                       // staged tile visible to all waves
        if (it + 1 < LQ / 64) issue_tile((it + 1) * 64, (it + 1) & 1);

        // ----- S = Q K^T for this wave's 16x64 strip -----
        v8f s[4] = {};
#pragma unroll
        for (int nt = 0; nt < 4; ++nt) {
            const __bf16* kp = &Klds[buf][nt * 16 + ln][0];
            t.half8[0] = *(const v8bf*)(kp + hi * 8);
            t.half8[1] = *(const v8bf*)(kp + 16 + hi * 8);
            v16bf b0 = t.v;                    // depth 0..31 slice
            t.half8[0] = *(const v8bf*)(kp + 32 + hi * 8);
            t.half8[1] = *(const v8bf*)(kp + 48 + hi * 8);
            v16bf b1 = t.v;                    // depth 32..63 slice
            s[nt] = wmma_bf16(qa0, b0, s[nt]);
            s[nt] = wmma_bf16(qa1, b1, s[nt]);
        }

        // ----- online softmax (row r lives in element r of a 16-lane half) -----
        float nm[8];
#pragma unroll
        for (int r = 0; r < 8; ++r) nm[r] = mrow[r];
#pragma unroll
        for (int nt = 0; nt < 4; ++nt)
#pragma unroll
            for (int r = 0; r < 8; ++r) {
                float v = s[nt][r] * scale;
                s[nt][r] = v;
                nm[r] = fmaxf(nm[r], v);
            }
#pragma unroll
        for (int mask = 1; mask < 16; mask <<= 1)
#pragma unroll
            for (int r = 0; r < 8; ++r)
                nm[r] = fmaxf(nm[r], __shfl_xor(nm[r], mask, 32));

        float ps[8];
#pragma unroll
        for (int r = 0; r < 8; ++r) ps[r] = 0.0f;
#pragma unroll
        for (int nt = 0; nt < 4; ++nt)
#pragma unroll
            for (int r = 0; r < 8; ++r) {
                float p = __expf(s[nt][r] - nm[r]);
                ps[r] += p;
                pbuf[w][hi * 8 + r][nt * 16 + ln] = (__bf16)p;
            }
#pragma unroll
        for (int mask = 1; mask < 16; mask <<= 1)
#pragma unroll
            for (int r = 0; r < 8; ++r)
                ps[r] += __shfl_xor(ps[r], mask, 32);

#pragma unroll
        for (int r = 0; r < 8; ++r) {
            float al = __expf(mrow[r] - nm[r]);
            lrow[r] = lrow[r] * al + ps[r];
            mrow[r] = nm[r];
#pragma unroll
            for (int nt = 0; nt < 4; ++nt) o[nt][r] *= al;
        }

        // wave-private LDS region: only need DS store->load ordering
        asm volatile("s_wait_dscnt 0x0" ::: "memory");

        // ----- reload P in A-layout, accumulate O += P * V -----
        v16bf pa0, pa1;
        t.half8[0] = *(const v8bf*)&pbuf[w][ln][hi * 8];
        t.half8[1] = *(const v8bf*)&pbuf[w][ln][16 + hi * 8];
        pa0 = t.v;
        t.half8[0] = *(const v8bf*)&pbuf[w][ln][32 + hi * 8];
        t.half8[1] = *(const v8bf*)&pbuf[w][ln][48 + hi * 8];
        pa1 = t.v;

#pragma unroll
        for (int nt = 0; nt < 4; ++nt) {
            const __bf16* vp = &Vlds[buf][nt * 16 + ln][0];
            t.half8[0] = *(const v8bf*)(vp + hi * 8);
            t.half8[1] = *(const v8bf*)(vp + 16 + hi * 8);
            v16bf vb0 = t.v;                   // keys kt..kt+31
            t.half8[0] = *(const v8bf*)(vp + 32 + hi * 8);
            t.half8[1] = *(const v8bf*)(vp + 48 + hi * 8);
            v16bf vb1 = t.v;                   // keys kt+32..kt+63
            o[nt] = wmma_bf16(pa0, vb0, o[nt]);
            o[nt] = wmma_bf16(pa1, vb1, o[nt]);
        }
        __syncthreads();   // everyone done with buf before it is refilled at it+2
    }

    // ----- epilogue: normalize, stage in LDS, coalesced f32 store -----
    float inv[8];
#pragma unroll
    for (int r = 0; r < 8; ++r) inv[r] = 1.0f / lrow[r];
#pragma unroll
    for (int nt = 0; nt < 4; ++nt)
#pragma unroll
        for (int r = 0; r < 8; ++r)
            otile[w * 16 + hi * 8 + r][nt * 16 + ln] = o[nt][r] * inv[r];
    __syncthreads();

    // out layout [B, L, H, 64]: 64 rows x 256B, 1024 chunks of 16B, 8/thread
#pragma unroll
    for (int i = 0; i < 8; ++i) {
        int c   = tid + i * 128;
        int row = c >> 4;
        int c4  = c & 15;
        float4 vv = *(const float4*)&otile[row][c4 * 4];
        size_t ob = (((size_t)b * LQ + mbase + row) * HQ + hd) * DEPTH + c4 * 4;
        *(float4*)(out + ob) = vv;
    }
}

// ---------------------------------------------------------------------------
extern "C" void kernel_launch(void* const* d_in, const int* in_sizes, int n_in,
                              void* d_out, int out_size, void* d_ws, size_t ws_size,
                              hipStream_t stream) {
    const float* query  = (const float*)d_in[0];
    const float* keys   = (const float*)d_in[1];
    const float* values = (const float*)d_in[2];
    const float* Wq     = (const float*)d_in[3];
    const float* Wk     = (const float*)d_in[4];
    const float* Wv     = (const float*)d_in[5];
    float* out = (float*)d_out;

    char* ws = (char*)d_ws;
    const size_t welems = (size_t)HQ * DEPTH * DMODEL;   // 1,048,576 per weight
    const size_t qelems = (size_t)BQ * HQ * LQ * DEPTH;  // 8,388,608 per tensor
    __bf16* Wqb = (__bf16*)(ws);
    __bf16* Wkb = Wqb + welems;
    __bf16* Wvb = Wkb + welems;
    __bf16* Qb  = Wvb + welems;
    __bf16* Kb  = Qb + qelems;
    __bf16* Vt  = Kb + qelems;
    // total ws use: (3*1M + 3*8.39M) bf16 ≈ 56.6 MB

    dim3 cb(256);
    dim3 cg((unsigned)((welems + 255) / 256));
    cvt_bf16_kernel<<<cg, cb, 0, stream>>>(Wq, Wqb, (int)welems);
    cvt_bf16_kernel<<<cg, cb, 0, stream>>>(Wk, Wkb, (int)welems);
    cvt_bf16_kernel<<<cg, cb, 0, stream>>>(Wv, Wvb, (int)welems);

    dim3 pg(LQ / 64, BQ * HQ);   // (16, 128)
    dim3 pb(128);                // 4 waves
    proj_kernel<<<pg, pb, 0, stream>>>(query,  Wqb, Qb, 0);
    proj_kernel<<<pg, pb, 0, stream>>>(keys,   Wkb, Kb, 0);
    proj_kernel<<<pg, pb, 0, stream>>>(values, Wvb, Vt, 1);

    attn_kernel<<<pg, pb, 0, stream>>>(Qb, Kb, Vt, out);
}
